// IMVTensorLSTM_22969485099269
// MI455X (gfx1250) — matrix operand order repaired and checked by reading, not compiled, not run on hardware
//
#include <hip/hip_runtime.h>
#include <stdint.h>

// ---------------------------------------------------------------------------
// IMV-Tensor-LSTM for MI455X (gfx1250, wave32, WMMA f32_16x16x32_f16)
//
// Kernel 1 (recurrent): grid (B/16, D) x 256 thr. Per block: one d, 16 batch
//   rows. Weight B-fragments (f16) register-resident across all T=256 steps;
//   h tile exchanged via LDS fp16; only 2 barriers per timestep. Attention
//   scalar exp(tanh(h.Fa+b)) computed via WMMA on wave 0, deferred one step
//   so it overlaps the gate WMMAs of all waves (reuses the same A fragments).
//   Softmax numerator/denominator accumulate on the fly -> no BxTxDxN tensor.
// Kernel 2 (head): normalizes alphas, computes mu/beta/mean.
// ---------------------------------------------------------------------------

typedef _Float16 v16h __attribute__((ext_vector_type(16)));
typedef float    v8f  __attribute__((ext_vector_type(8)));
typedef _Float16 h2   __attribute__((ext_vector_type(2)));

union Frag16 { v16h v; uint32_t u[8]; };

constexpr int Bc = 64, Tc = 256, Dc = 32, Nc = 128, MT = 16;

__device__ inline float fast_rcp(float x) { return __builtin_amdgcn_rcpf(x); }
// sigmoid(x) = 1/(1+e^-x); tanh(x) = 1 - 2/(1+e^{2x})  (division-free, v_rcp)
__device__ inline float fast_sigmoid(float x) { return fast_rcp(1.0f + __expf(-x)); }
__device__ inline float fast_tanh(float x)    { return 1.0f - 2.0f * fast_rcp(1.0f + __expf(2.0f * x)); }
__device__ inline uint32_t pack_h2(float a, float b) {
    h2 p; p[0] = (_Float16)a; p[1] = (_Float16)b;
    return *(const uint32_t*)&p;
}
__device__ inline v8f wmma_f16(const Frag16& a, const Frag16& b, v8f c) {
    return __builtin_amdgcn_wmma_f32_16x16x32_f16(false, a.v, false, b.v,
                                                  (short)0, c, false, false);
}

__global__ __launch_bounds__(256) void imv_lstm_recurrent(
    const float* __restrict__ x,
    const float* __restrict__ Uj, const float* __restrict__ Ui,
    const float* __restrict__ Uf, const float* __restrict__ Uo,
    const float* __restrict__ Wj, const float* __restrict__ Wi,
    const float* __restrict__ Wf, const float* __restrict__ Wo,
    const float* __restrict__ bj, const float* __restrict__ bi,
    const float* __restrict__ bf, const float* __restrict__ bo,
    const float* __restrict__ Fa, const float* __restrict__ Fab,
    float* __restrict__ alpha_raw,   // (B,T,D) unnormalized exp(tanh(.))
    float* __restrict__ den_out,     // (B,D)   sum over t of alpha_raw
    float* __restrict__ h_last_out,  // (B,D,N)
    float* __restrict__ num_out)     // (B,D,N) sum over t of alpha_raw*h_t
{
    const int b0   = blockIdx.x * MT;   // batch tile base (0,16,32,48)
    const int d    = blockIdx.y;        // variable index 0..31
    const int tid  = threadIdx.x;
    const int lane = tid & 31;
    const int wv   = tid >> 5;          // wave 0..7 -> N tile
    const int n0   = wv * 16;
    const int nl   = lane & 15;
    const int half = lane >> 4;

    __shared__ _Float16 hLDS[MT * Nc];  // state h, fp16, row-major 16x128
    __shared__ float xtLDS[MT];
    __shared__ float alphaLDS[MT];

    for (int i = tid; i < MT * Nc; i += 256) hLDS[i] = (_Float16)0.0f;
    const float ab = Fab[d];

    // ---- loop-invariant per-lane U / bias values (registers) ----
    float ureg[4], breg[4];
    {
        const float* Ug[4] = { Uj, Ui, Uf, Uo };
        const float* bg[4] = { bj, bi, bf, bo };
        #pragma unroll
        for (int g = 0; g < 4; ++g) {
            ureg[g] = Ug[g][d * Nc + n0 + nl];
            breg[g] = bg[g][d * Nc + n0 + nl];
        }
    }

    // ---- register-resident WMMA B fragments (weights, fp32 -> fp16) ----
    // B layout used here (16-bit 32x16 tile, contiguous k order, same as A): lanes 0-15 hold K=0..15,
    // lanes 16-31 hold K=16..31; VGPR v packs rows k=half*16+2v, +1.
    Frag16 bfrag[4][4];
    const float* const Wg[4] = { Wj, Wi, Wf, Wo };
    #pragma unroll
    for (int g = 0; g < 4; ++g) {
        const float* W = Wg[g] + (size_t)d * Nc * Nc; // W[d][k][n], k=contraction
        #pragma unroll
        for (int s = 0; s < 4; ++s) {
            #pragma unroll
            for (int v = 0; v < 8; ++v) {
                int k = s * 32 + half * 16 + v * 2;
                bfrag[g][s].u[v] = pack_h2(W[(k + 0) * Nc + n0 + nl],
                                           W[(k + 1) * Nc + n0 + nl]);
            }
        }
    }

    // ---- F_alpha packed into column 0 of a B fragment (for adot WMMA) ----
    Frag16 fafrag[4];
    #pragma unroll
    for (int s = 0; s < 4; ++s) {
        #pragma unroll
        for (int v = 0; v < 8; ++v) {
            int k = s * 32 + half * 16 + v * 2;
            float f0 = (nl == 0) ? Fa[d * Nc + k]     : 0.0f;
            float f1 = (nl == 0) ? Fa[d * Nc + k + 1] : 0.0f;
            fafrag[s].u[v] = pack_h2(f0, f1);
        }
    }

    v8f cst, num, hnew, hprev;
    #pragma unroll
    for (int r = 0; r < 8; ++r) { cst[r] = 0.0f; num[r] = 0.0f; hnew[r] = 0.0f; hprev[r] = 0.0f; }
    float den8[8];
    #pragma unroll
    for (int r = 0; r < 8; ++r) den8[r] = 0.0f;

    __syncthreads();

    for (int t = 0; t < Tc; ++t) {
        if (tid < MT) xtLDS[tid] = x[((size_t)(b0 + tid) * Tc + t) * Dc + d];
        __syncthreads();   // sync A: x_t and h state (h^t) visible

        // ---- A fragments (state h^t) from LDS, contiguous-k 16x32 A layout (same k order as B) ----
        Frag16 afrag[4];
        #pragma unroll
        for (int s = 0; s < 4; ++s) {
            #pragma unroll
            for (int v = 0; v < 8; ++v) {
                int kk = s * 32 + half * 16 + v * 2;
                afrag[s].u[v] = *(const uint32_t*)&hLDS[(lane & 15) * Nc + kk];
            }
        }
        float xv[8];
        #pragma unroll
        for (int r = 0; r < 8; ++r) xv[r] = xtLDS[r + half * 8];

        // ---- deferred attention scalar for output h^t (i.e. index t-1),
        //      overlapped with everyone's gate WMMAs (wave 0 only) ----
        if (wv == 0 && t > 0) {
            v8f ad;
            #pragma unroll
            for (int r = 0; r < 8; ++r) ad[r] = ab;
            #pragma unroll
            for (int s = 0; s < 4; ++s) ad = wmma_f16(afrag[s], fafrag[s], ad);
            if (nl == 0) {   // lanes 0 (rows 0-7) and 16 (rows 8-15) hold col 0
                #pragma unroll
                for (int r = 0; r < 8; ++r) {
                    int mrow = r + half * 8;
                    float ar = __expf(fast_tanh(ad[r]));
                    alphaLDS[mrow] = ar;
                    den8[r] += ar;
                    alpha_raw[((size_t)(b0 + mrow) * Tc + (t - 1)) * Dc + d] = ar;
                }
            }
        }

        // ---- four gates with running combine: j -> ij -> c -> h ----
        v8f acc, run;
        #pragma unroll
        for (int r = 0; r < 8; ++r) acc[r] = xv[r] * ureg[0] + breg[0];
        #pragma unroll
        for (int s = 0; s < 4; ++s) acc = wmma_f16(afrag[s], bfrag[0][s], acc);
        #pragma unroll
        for (int r = 0; r < 8; ++r) run[r] = fast_tanh(acc[r]);           // j

        #pragma unroll
        for (int r = 0; r < 8; ++r) acc[r] = xv[r] * ureg[1] + breg[1];
        #pragma unroll
        for (int s = 0; s < 4; ++s) acc = wmma_f16(afrag[s], bfrag[1][s], acc);
        #pragma unroll
        for (int r = 0; r < 8; ++r) run[r] *= fast_sigmoid(acc[r]);       // i*j

        #pragma unroll
        for (int r = 0; r < 8; ++r) acc[r] = xv[r] * ureg[2] + breg[2];
        #pragma unroll
        for (int s = 0; s < 4; ++s) acc = wmma_f16(afrag[s], bfrag[2][s], acc);
        #pragma unroll
        for (int r = 0; r < 8; ++r)
            cst[r] = cst[r] * fast_sigmoid(acc[r]) + run[r];              // c

        #pragma unroll
        for (int r = 0; r < 8; ++r) acc[r] = xv[r] * ureg[3] + breg[3];
        #pragma unroll
        for (int s = 0; s < 4; ++s) acc = wmma_f16(afrag[s], bfrag[3][s], acc);
        #pragma unroll
        for (int r = 0; r < 8; ++r)
            hnew[r] = fast_sigmoid(acc[r]) * fast_tanh(cst[r]);           // h^{t+1}

        __syncthreads();   // sync B: hLDS reads done, alphaLDS visible

        if (t > 0) {
            #pragma unroll
            for (int r = 0; r < 8; ++r)
                num[r] += alphaLDS[r + half * 8] * hprev[r];   // hprev = h^t
        }
        #pragma unroll
        for (int r = 0; r < 8; ++r) {
            hLDS[(r + half * 8) * Nc + n0 + nl] = (_Float16)hnew[r];
            hprev[r] = hnew[r];
        }
    }

    // ---- epilogue: attention for the final output h^Tc (index Tc-1) ----
    __syncthreads();       // h^Tc visible in hLDS
    if (wv == 0) {
        Frag16 afrag[4];
        #pragma unroll
        for (int s = 0; s < 4; ++s) {
            #pragma unroll
            for (int v = 0; v < 8; ++v) {
                int kk = s * 32 + half * 16 + v * 2;
                afrag[s].u[v] = *(const uint32_t*)&hLDS[(lane & 15) * Nc + kk];
            }
        }
        v8f ad;
        #pragma unroll
        for (int r = 0; r < 8; ++r) ad[r] = ab;
        #pragma unroll
        for (int s = 0; s < 4; ++s) ad = wmma_f16(afrag[s], fafrag[s], ad);
        if (nl == 0) {
            #pragma unroll
            for (int r = 0; r < 8; ++r) {
                int mrow = r + half * 8;
                float ar = __expf(fast_tanh(ad[r]));
                alphaLDS[mrow] = ar;
                den8[r] += ar;
                alpha_raw[((size_t)(b0 + mrow) * Tc + (Tc - 1)) * Dc + d] = ar;
            }
        }
    }
    __syncthreads();
    #pragma unroll
    for (int r = 0; r < 8; ++r)
        num[r] += alphaLDS[r + half * 8] * hnew[r];

    if (wv == 0 && nl == 0) {
        #pragma unroll
        for (int r = 0; r < 8; ++r)
            den_out[(b0 + r + half * 8) * Dc + d] = den8[r];
    }
    #pragma unroll
    for (int r = 0; r < 8; ++r) {
        size_t o = ((size_t)(b0 + r + half * 8) * Dc + d) * Nc + n0 + nl;
        h_last_out[o] = hnew[r];
        num_out[o]    = num[r];
    }
}

__global__ __launch_bounds__(256) void imv_lstm_head(
    const float* __restrict__ Fbw, const float* __restrict__ Fbb,
    const float* __restrict__ Pw,  const float* __restrict__ Pb,
    const float* __restrict__ alpha_raw, const float* __restrict__ den,
    const float* __restrict__ h_last,    const float* __restrict__ num,
    float* __restrict__ out)  // [ mean(B) | alphas(B*T*D) | betas(B*D) ]
{
    const int b = blockIdx.x;
    const int tid = threadIdx.x;
    __shared__ float invdenL[Dc], muL[Dc], brL[Dc];

    if (tid < Dc) invdenL[tid] = fast_rcp(den[b * Dc + tid]);
    __syncthreads();

    if (tid < Dc) {
        const int d = tid;
        const float invden = invdenL[d];
        float mu = Pb[0];
        float bp = Fbb[0];
        for (int n = 0; n < Nc; ++n) {
            size_t o = ((size_t)b * Dc + d) * Nc + n;
            float gn = num[o] * invden;   // attention-pooled g_n
            float hl = h_last[o];
            mu += gn * Pw[n]  + hl * Pw[Nc + n];
            bp += gn * Fbw[n] + hl * Fbw[Nc + n];
        }
        muL[d] = mu;
        brL[d] = __expf(fast_tanh(bp));
    }
    __syncthreads();

    if (tid == 0) {
        float s = 0.0f;
        for (int d = 0; d < Dc; ++d) s += brL[d];
        const float inv = fast_rcp(s);
        float mean = 0.0f;
        for (int d = 0; d < Dc; ++d) {
            float beta = brL[d] * inv;
            out[Bc + (size_t)Bc * Tc * Dc + (size_t)b * Dc + d] = beta;
            mean += beta * muL[d];
        }
        out[b] = mean;
    }

    // normalized alphas, contiguous (t*D + d) per batch
    for (int idx = tid; idx < Tc * Dc; idx += 256) {
        int dd = idx % Dc;
        out[Bc + (size_t)b * Tc * Dc + idx] =
            alpha_raw[(size_t)b * Tc * Dc + idx] * invdenL[dd];
    }
}

extern "C" void kernel_launch(void* const* d_in, const int* in_sizes, int n_in,
                              void* d_out, int out_size, void* d_ws, size_t ws_size,
                              hipStream_t stream) {
    const float* x   = (const float*)d_in[0];
    const float* U_j = (const float*)d_in[1];
    const float* U_i = (const float*)d_in[2];
    const float* U_f = (const float*)d_in[3];
    const float* U_o = (const float*)d_in[4];
    const float* W_j = (const float*)d_in[5];
    const float* W_i = (const float*)d_in[6];
    const float* W_f = (const float*)d_in[7];
    const float* W_o = (const float*)d_in[8];
    const float* b_j = (const float*)d_in[9];
    const float* b_i = (const float*)d_in[10];
    const float* b_f = (const float*)d_in[11];
    const float* b_o = (const float*)d_in[12];
    const float* F_alpha_n   = (const float*)d_in[13];
    const float* F_alpha_n_b = (const float*)d_in[14];
    const float* F_beta_w = (const float*)d_in[15];
    const float* F_beta_b = (const float*)d_in[16];
    const float* Phi_w    = (const float*)d_in[17];
    const float* Phi_b    = (const float*)d_in[18];

    float* wsf       = (float*)d_ws;
    float* alpha_raw = wsf;                                  // B*T*D = 524288
    float* den       = alpha_raw + (size_t)Bc * Tc * Dc;     // B*D    = 2048
    float* h_last    = den + (size_t)Bc * Dc;                // B*D*N  = 262144
    float* num       = h_last + (size_t)Bc * Dc * Nc;        // B*D*N  = 262144

    dim3 grid1(Bc / MT, Dc);  // (4, 32)
    imv_lstm_recurrent<<<grid1, 256, 0, stream>>>(
        x, U_j, U_i, U_f, U_o, W_j, W_i, W_f, W_o,
        b_j, b_i, b_f, b_o, F_alpha_n, F_alpha_n_b,
        alpha_raw, den, h_last, num);

    imv_lstm_head<<<Bc, 256, 0, stream>>>(
        F_beta_w, F_beta_b, Phi_w, Phi_b,
        alpha_raw, den, h_last, num, (float*)d_out);
}
